// nat_base_54735063220687
// MI455X (gfx1250) — matrix operand order repaired and checked by reading, not compiled, not run on hardware
//
#include <hip/hip_runtime.h>
#include <cmath>

// ---------------------------------------------------------------------------
// NAT-base forward for MI455X (gfx1250, wave32, WMMA).
// GEMMs (QKV/proj/FC1/FC2 + 3x3 convs as implicit GEMM) -> v_wmma_f32_16x16x32_f16
// with f32 accumulate, branchless vectorized staging (global_load_b128 ->
// cvt_pk -> ds_store_b128), software-pipelined K loop, fused bias/GELU/residual
// epilogue. Vector-staging legality is a compile-time template parameter so the
// hot GEMMs carry zero divergent staging branches; only conv1 (K=27) uses the
// clamped-scalar path. LayerNorm + 49-tap clamped neighborhood attention stay
// on VALU (wave32, hd==32 -> one lane per channel).
// ---------------------------------------------------------------------------

typedef _Float16 half_t;
typedef __attribute__((ext_vector_type(16))) _Float16 v16h;
typedef __attribute__((ext_vector_type(8)))  _Float16 v8h;
typedef __attribute__((ext_vector_type(8)))  float    v8f;
typedef __attribute__((ext_vector_type(4)))  float    v4f;

#define TILE_M 64
#define TILE_N 64
#define TILE_K 32
#define LDSK   40   // halves; row stride 80B = 5*16B -> b128-aligned chunks

__device__ __forceinline__ float gelu_exact(float v) {
    return 0.5f * v * (1.0f + erff(v * 0.70710678118654752f));
}

// 8-wide row-major fetch.
// KVEC=true : caller guarantees Kd % 32 == 0 and k+8 <= Kd -> unconditional
//             2x global_load_b128 (rows are 16B aligned), row clamped + masked.
// KVEC=false: clamped scalar loads, OOB-K masked (conv1 weights, K=27).
template<bool KVEC>
__device__ __forceinline__ void fetch8_rowmajor(const float* __restrict__ P,
                                                int rows, int Kd, int r, int k,
                                                float out[8])
{
    const int rc = r < rows ? r : rows - 1;
    const bool rok = r < rows;
    if (KVEC) {
        const v4f* p = (const v4f*)(P + (size_t)rc * Kd + k);
        const v4f u0 = p[0];
        const v4f u1 = p[1];
        #pragma unroll
        for (int j = 0; j < 4; ++j) out[j]     = rok ? u0[j] : 0.0f;
        #pragma unroll
        for (int j = 0; j < 4; ++j) out[4 + j] = rok ? u1[j] : 0.0f;
    } else {
        #pragma unroll
        for (int j = 0; j < 8; ++j) {
            const int kc = (k + j) < Kd ? (k + j) : Kd - 1;
            const float v = P[(size_t)rc * Kd + kc];
            out[j] = (rok && (k + j) < Kd) ? v : 0.0f;
        }
    }
}

// Branchless implicit-im2col fetch for 3x3 stride-s pad-1 conv (gather; always
// scalar loads with clamped addresses + select-zero masks, no exec branches).
__device__ __forceinline__ void fetch8_conv(const float* __restrict__ A,
                                            int M, int Kd, int m, int k,
                                            int inH, int inW, int Cin,
                                            int outH, int outW, int stride, int nchw,
                                            float out[8])
{
    const int mc = m < M ? m : M - 1;
    const int ow = mc % outW;
    const int t1 = mc / outW;
    const int oh = t1 % outH;
    const int b  = t1 / outH;
    const bool mok = m < M;
    #pragma unroll
    for (int j = 0; j < 8; ++j) {
        const int kj = k + j;
        const int kc = kj < Kd ? kj : Kd - 1;
        const int ci = kc / 9;
        const int r9 = kc - ci * 9;
        const int ky = r9 / 3;
        const int kx = r9 - ky * 3;
        const int ih = oh * stride - 1 + ky;
        const int iw = ow * stride - 1 + kx;
        const int ihc = min(max(ih, 0), inH - 1);
        const int iwc = min(max(iw, 0), inW - 1);
        const float v = nchw
            ? A[(((size_t)b * Cin + ci) * inH + ihc) * inW + iwc]
            : A[(((size_t)b * inH + ihc) * inW + iwc) * Cin + ci];
        const bool ok = mok && (kj < Kd) && (ih == ihc) && (iw == iwc);
        out[j] = ok ? v : 0.0f;
    }
}

// ---------------------------------------------------------------------------
// Fused GEMM:  Y[M,N] = act( A[M,K] * W[N,K]^T + bias ) (+ resid)
// Block: 256 threads = 8 waves; 64x64 tile; each wave owns a 16x32 slab
// (two v_wmma_f32_16x16x32_f16 per K step). Next K chunk is prefetched into
// registers while the WMMAs run. N is always a multiple of 64 here.
// ---------------------------------------------------------------------------
template<bool CONV, bool KVEC>
__global__ __launch_bounds__(256)
void gemm_wmma_kernel(const float* __restrict__ A, const float* __restrict__ Wt,
                      const float* __restrict__ bias, const float* __restrict__ resid,
                      float* __restrict__ Y, int M, int N, int Kd, int act,
                      int inH, int inW, int Cin,
                      int outH, int outW, int cstride, int nchw)
{
    __shared__ half_t sA[TILE_M][LDSK];
    __shared__ half_t sB[TILE_N][LDSK];

    const int tid = threadIdx.x;
    const int m0  = blockIdx.y * TILE_M;
    const int n0  = blockIdx.x * TILE_N;

    const int lrow = tid >> 2;        // 0..63 staged row
    const int lk   = (tid & 3) * 8;   // 0,8,16,24

    const int wave = tid >> 5;
    const int lane = tid & 31;
    const int mt   = wave & 3;        // 16-row sub-tile
    const int ntp  = wave >> 2;       // 32-col half

    v8f acc0 = {};
    v8f acc1 = {};

    // warm the L2/GL1 path for the residual rows this block will consume
    if (resid) __builtin_prefetch(resid + (size_t)(m0 + mt * 16) * N + n0, 0, 1);

    float ar[8], wr[8];
    if (CONV) fetch8_conv(A, M, Kd, m0 + lrow, lk, inH, inW, Cin, outH, outW, cstride, nchw, ar);
    else      fetch8_rowmajor<KVEC>(A, M, Kd, m0 + lrow, lk, ar);
    fetch8_rowmajor<KVEC>(Wt, N, Kd, n0 + lrow, lk, wr);

    for (int k0 = 0; k0 < Kd; k0 += TILE_K) {
        // ---- commit prefetched chunk to LDS as packed f16 (ds_store_b128) ----
        v8h pa, pw;
        #pragma unroll
        for (int j = 0; j < 8; ++j) { pa[j] = (half_t)ar[j]; pw[j] = (half_t)wr[j]; }
        *(v8h*)&sA[lrow][lk] = pa;
        *(v8h*)&sB[lrow][lk] = pw;
        __syncthreads();

        // ---- prefetch next K chunk while WMMAs run ----
        // (guard k0+32 < Kd with Kd%32==0 implies kn+8 <= Kd for KVEC path)
        const int kn = k0 + TILE_K + lk;
        if (k0 + TILE_K < Kd) {
            if (CONV) fetch8_conv(A, M, Kd, m0 + lrow, kn, inH, inW, Cin, outH, outW, cstride, nchw, ar);
            else      fetch8_rowmajor<KVEC>(A, M, Kd, m0 + lrow, kn, ar);
            fetch8_rowmajor<KVEC>(Wt, N, Kd, n0 + lrow, kn, wr);
        }

        // ---- A fragment (contiguous K per lane half, matching B) ----
        v16h afrag, b0, b1;
        const int am = mt * 16 + (lane & 15);
        const int kb = (lane >> 4) * 16;
        #pragma unroll
        for (int h = 0; h < 16; ++h) {
            const int kk = kb + h;
            afrag[h] = sA[am][kk];
        }
        // ---- B fragments: lane L -> column N = L&15, K half by L>=16 ----
        const int koff = (lane >> 4) * 16;
        const int bn0  = ntp * 32 + (lane & 15);
        #pragma unroll
        for (int h = 0; h < 16; ++h) b0[h] = sB[bn0][koff + h];
        #pragma unroll
        for (int h = 0; h < 16; ++h) b1[h] = sB[bn0 + 16][koff + h];

        acc0 = __builtin_amdgcn_wmma_f32_16x16x32_f16(false, afrag, false, b0,
                                                      (short)0, acc0, false, false);
        acc1 = __builtin_amdgcn_wmma_f32_16x16x32_f16(false, afrag, false, b1,
                                                      (short)0, acc1, false, false);
        __syncthreads();
    }

    // ---- epilogue: bias (+GELU) (+residual); N % 64 == 0 so no n guard ----
    const int nl = lane & 15;
    const int mh = (lane >> 4) * 8;
    #pragma unroll
    for (int t = 0; t < 2; ++t) {
        const v8f acc = t ? acc1 : acc0;
        const int n = n0 + ntp * 32 + t * 16 + nl;
        const float bn = bias[n];
        #pragma unroll
        for (int r = 0; r < 8; ++r) {
            const int m = m0 + mt * 16 + mh + r;
            if (m < M) {
                float v = acc[r] + bn;
                if (act) v = gelu_exact(v);
                if (resid) v += resid[(size_t)m * N + n];
                Y[(size_t)m * N + n] = v;
            }
        }
    }
}

// ---------------------------------------------------------------------------
// LayerNorm: one wave32 per row; C in {128, 256}.
// ---------------------------------------------------------------------------
__global__ __launch_bounds__(256)
void ln_kernel(const float* __restrict__ x, const float* __restrict__ g,
               const float* __restrict__ b, float* __restrict__ y, int M, int C)
{
    const int wave = threadIdx.x >> 5;
    const int lane = threadIdx.x & 31;
    const int row  = blockIdx.x * 8 + wave;
    if (row >= M) return;

    const float* xr = x + (size_t)row * C;
    const int nch = C >> 5;     // <= 8
    float vals[8];
    float s = 0.0f, s2 = 0.0f;
    for (int j = 0; j < nch; ++j) {
        const float v = xr[lane + j * 32];
        vals[j] = v; s += v; s2 += v * v;
    }
    #pragma unroll
    for (int off = 16; off > 0; off >>= 1) {
        s  += __shfl_xor(s,  off, 32);
        s2 += __shfl_xor(s2, off, 32);
    }
    const float mu  = s / (float)C;
    const float var = s2 / (float)C - mu * mu;
    const float rs  = rsqrtf(var + 1e-5f);
    float* yr = y + (size_t)row * C;
    for (int j = 0; j < nch; ++j) {
        const int c = lane + j * 32;
        yr[c] = (vals[j] - mu) * rs * g[c] + b[c];
    }
}

// ---------------------------------------------------------------------------
// Neighborhood attention (K=7, boundary-clamped), hd = C/heads = 32.
// One wave32 per (pixel, head). q cached in 8 v4f registers; k/v rows are
// 128B-aligned so dots use global_load_b128. Softmax staged through LDS.
// ---------------------------------------------------------------------------
__global__ __launch_bounds__(256)
void natten_kernel(const float* __restrict__ qkv, const float* __restrict__ rpb,
                   float* __restrict__ out, int Bn, int H, int W, int C, int heads)
{
    __shared__ float sAttn[8][64];
    const int wave  = threadIdx.x >> 5;
    const int lane  = threadIdx.x & 31;
    const int total = Bn * H * W * heads;
    int gw = blockIdx.x * 8 + wave;
    const bool active = gw < total;
    if (!active) gw = total - 1;           // uniform barrier participation

    const int hd  = C / heads;             // 32
    const int e   = gw % heads;
    const int pix = gw / heads;
    const int px  = pix % W;
    const int t1  = pix / W;
    const int py  = t1 % H;
    const int b   = t1 / H;

    const int sh = min(max(py - 3, 0), H - 7);
    const int sw = min(max(px - 3, 0), W - 7);
    const float scale = rsqrtf((float)hd);

    const v4f* q4 = (const v4f*)(qkv + (size_t)pix * 3 * C + e * hd);
    v4f qr[8];
    #pragma unroll
    for (int d = 0; d < 8; ++d) qr[d] = q4[d];

    // scores: lane n handles neighbors n and n+32
    for (int n = lane; n < 49; n += 32) {
        const int ky = n / 7, kx = n - ky * 7;
        const int iy = sh + ky, ix = sw + kx;
        const int pix2 = (b * H + iy) * W + ix;
        const v4f* k4 = (const v4f*)(qkv + (size_t)pix2 * 3 * C + C + e * hd);
        float s = 0.0f;
        #pragma unroll
        for (int d = 0; d < 8; ++d) {
            const v4f kv = k4[d];
            s += qr[d][0] * kv[0] + qr[d][1] * kv[1] + qr[d][2] * kv[2] + qr[d][3] * kv[3];
        }
        const int rh = iy - py + 6, rw = ix - px + 6;
        sAttn[wave][n] = s * scale + rpb[(e * 13 + rh) * 13 + rw];
    }
    __syncthreads();

    // softmax over 49
    float v0 = sAttn[wave][lane];
    float v1 = (lane + 32 < 49) ? sAttn[wave][lane + 32] : -3.4e38f;
    float mx = fmaxf(v0, v1);
    #pragma unroll
    for (int off = 16; off > 0; off >>= 1) mx = fmaxf(mx, __shfl_xor(mx, off, 32));
    const float e0 = __expf(v0 - mx);
    const float e1 = (lane + 32 < 49) ? __expf(v1 - mx) : 0.0f;
    float sum = e0 + e1;
    #pragma unroll
    for (int off = 16; off > 0; off >>= 1) sum += __shfl_xor(sum, off, 32);
    const float inv = 1.0f / sum;
    sAttn[wave][lane] = e0 * inv;
    if (lane + 32 < 49) sAttn[wave][lane + 32] = e1 * inv;
    __syncthreads();

    // AV: lane = head-dim channel (lanes load consecutive floats -> coalesced)
    float acc = 0.0f;
    for (int n = 0; n < 49; ++n) {
        const int ky = n / 7, kx = n - ky * 7;
        const int pix2 = (b * H + sh + ky) * W + sw + kx;
        acc += sAttn[wave][n] * qkv[(size_t)pix2 * 3 * C + 2 * C + e * hd + lane];
    }
    if (active) out[(size_t)pix * C + e * hd + lane] = acc;
}

// ---------------------------------------------------------------------------
// Final NHWC -> NCHW transpose into d_out.
// ---------------------------------------------------------------------------
__global__ __launch_bounds__(256)
void nhwc_to_nchw_kernel(const float* __restrict__ in, float* __restrict__ out,
                         int B, int H, int W, int C)
{
    const int i = blockIdx.x * 256 + threadIdx.x;
    const int total = B * H * W * C;
    if (i >= total) return;
    const int c = i % C;
    int t = i / C;
    const int w = t % W; t /= W;
    const int h = t % H;
    const int b = t / H;
    out[(((size_t)b * C + c) * H + h) * W + w] = in[i];
}

// ---------------------------------------------------------------------------
extern "C" void kernel_launch(void* const* d_in, const int* in_sizes, int n_in,
                              void* d_out, int out_size, void* d_ws, size_t ws_size,
                              hipStream_t stream)
{
    (void)in_sizes; (void)n_in; (void)out_size; (void)ws_size;

    const float* x     = (const float*)d_in[0];
    const float* pe_w1 = (const float*)d_in[1];
    const float* pe_b1 = (const float*)d_in[2];
    const float* pe_w2 = (const float*)d_in[3];
    const float* pe_b2 = (const float*)d_in[4];
    const float* pe_g  = (const float*)d_in[5];
    const float* pe_be = (const float*)d_in[6];
    const float* L0[13];
    for (int i = 0; i < 13; ++i) L0[i] = (const float*)d_in[7 + i];
    const float* ds_w  = (const float*)d_in[20];
    const float* ds_b  = (const float*)d_in[21];
    const float* ds_g  = (const float*)d_in[22];
    const float* ds_be = (const float*)d_in[23];
    const float* L1[13];
    for (int i = 0; i < 13; ++i) L1[i] = (const float*)d_in[24 + i];

    // workspace (fp32): everything L2-resident (~32 MB total)
    float* f0 = (float*)d_ws;       // conv1 out [2,112,112,64] ; later x1 [2,28,28,256]
    float* f1 = f0 + 1605632;       // x  (level 0)  [6272,128]
    float* f2 = f1 + 802816;        // LN scratch h
    float* f3 = f2 + 802816;        // qkv   (max 6272*384)
    float* f4 = f3 + 2408448;       // attention out
    float* f5 = f4 + 802816;        // mlp hidden (max 6272*256)

    auto gemm = [&](const float* A, const float* Wt, const float* bias,
                    const float* resid, float* Y, int M, int N, int K, int act) {
        dim3 grid(N / 64, (M + 63) / 64);
        gemm_wmma_kernel<false, true><<<grid, 256, 0, stream>>>(
            A, Wt, bias, resid, Y, M, N, K, act, 0, 0, 0, 0, 0, 0, 0);
    };
    auto conv = [&](const float* A, const float* Wt, const float* bias, float* Y,
                    int N, int K, int B, int inH, int inW, int Cin,
                    int outH, int outW, int stride, int nchw) {
        const int M = B * outH * outW;
        dim3 grid(N / 64, (M + 63) / 64);
        if ((K & 31) == 0) {
            gemm_wmma_kernel<true, true><<<grid, 256, 0, stream>>>(
                A, Wt, bias, nullptr, Y, M, N, K, 0,
                inH, inW, Cin, outH, outW, stride, nchw);
        } else {  // conv1: K=27, unaligned weight rows -> clamped-scalar staging
            gemm_wmma_kernel<true, false><<<grid, 256, 0, stream>>>(
                A, Wt, bias, nullptr, Y, M, N, K, 0,
                inH, inW, Cin, outH, outW, stride, nchw);
        }
    };

    auto run_level = [&](const float* const* P, float* xbuf, int nblocks,
                         int C, int heads, int B, int H, int W) {
        const int M  = B * H * W;
        const int C3 = 3 * C, C2 = 2 * C;
        for (int i = 0; i < nblocks; ++i) {
            ln_kernel<<<M / 8, 256, 0, stream>>>(xbuf, P[0] + (size_t)i * C,
                                                 P[1] + (size_t)i * C, f2, M, C);
            gemm(f2, P[2] + (size_t)i * C3 * C, P[3] + (size_t)i * C3,
                 nullptr, f3, M, C3, C, 0);
            natten_kernel<<<(M * heads) / 8, 256, 0, stream>>>(
                f3, P[4] + (size_t)i * heads * 169, f4, B, H, W, C, heads);
            gemm(f4, P[5] + (size_t)i * C * C, P[6] + (size_t)i * C,
                 xbuf, xbuf, M, C, C, 0);
            ln_kernel<<<M / 8, 256, 0, stream>>>(xbuf, P[7] + (size_t)i * C,
                                                 P[8] + (size_t)i * C, f2, M, C);
            gemm(f2, P[9] + (size_t)i * C2 * C, P[10] + (size_t)i * C2,
                 nullptr, f5, M, C2, C, 1 /*GELU*/);
            gemm(f5, P[11] + (size_t)i * C * C2, P[12] + (size_t)i * C,
                 xbuf, xbuf, M, C, C2, 0);
        }
    };

    // ---- patch embed: conv 3->64 s2 (NCHW in), conv 64->128 s2, LN ----
    conv(x,  pe_w1, pe_b1, f0,  64,   27, 2, 224, 224,  3, 112, 112, 2, 1);
    conv(f0, pe_w2, pe_b2, f1, 128,  576, 2, 112, 112, 64,  56,  56, 2, 0);
    ln_kernel<<<6272 / 8, 256, 0, stream>>>(f1, pe_g, pe_be, f1, 6272, 128);

    // ---- level 0: 3 blocks @ 56x56, C=128, heads=4 ----
    run_level(L0, f1, 3, 128, 4, 2, 56, 56);

    // ---- downsample: conv 128->256 s2 + LN -> [2,28,28,256] ----
    conv(f1, ds_w, ds_b, f0, 256, 1152, 2, 56, 56, 128, 28, 28, 2, 0);
    ln_kernel<<<1568 / 8, 256, 0, stream>>>(f0, ds_g, ds_be, f0, 1568, 256);

    // ---- level 1: 4 blocks @ 28x28, C=256, heads=8 ----
    run_level(L1, f0, 4, 256, 8, 2, 28, 28);

    // ---- output: NHWC -> NCHW [2,256,28,28] ----
    nhwc_to_nchw_kernel<<<(401408 + 255) / 256, 256, 0, stream>>>(
        f0, (float*)d_out, 2, 28, 28, 256);
}
